// Adapter_CrossNonParam_60146722013423
// MI455X (gfx1250) — matrix operand order repaired and checked by reading, not compiled, not run on hardware
//
#include <hip/hip_runtime.h>
#include <hip/hip_bf16.h>
#include <math.h>

// MI455X / gfx1250: wave32, WMMA 16x16x32 bf16 path.
typedef __bf16 bf16;
typedef __attribute__((ext_vector_type(16))) __bf16 v16bf;
typedef __attribute__((ext_vector_type(8)))  __bf16 v8bf;
typedef __attribute__((ext_vector_type(8)))  float  v8f;

#define WMMA_BF16(a, b, c) \
  __builtin_amdgcn_wmma_f32_16x16x32_bf16(false, (a), false, (b), (short)0, (c), false, false)

constexpr int   B_ = 32, N_ = 2248, C_ = 1024, D_ = 128, P_ = 200, T_ = 2048;
constexpr float SCALE_ = 0.08838834764831845f; // 128^-0.5

static __device__ __forceinline__ v16bf cat8(v8bf a, v8bf b) {
  return __builtin_shufflevector(a, b, 0,1,2,3,4,5,6,7,8,9,10,11,12,13,14,15);
}

// ---------------------------------------------------------------------------
// Kernel 0: pre-swizzle W_down / W_up (fp32 -> bf16) into WMMA B-fragment
// layout: [kstep][tile][lane][slot], slot s of lane L holds K = (s&7) + 8*(L>=16) + 16*(s>=8).
// ---------------------------------------------------------------------------
__global__ void k_swizzle_weights(const float* __restrict__ Wd,
                                  const float* __restrict__ Wu,
                                  bf16* __restrict__ Wd_sw,
                                  bf16* __restrict__ Wu_sw) {
  int i = blockIdx.x * blockDim.x + threadIdx.x;
  if (i < 32 * 8 * 32 * 16) {              // W_down: 32 ksteps x 8 d-tiles
    int s = i & 15, L = (i >> 4) & 31, t = (i >> 9) & 7, kk = i >> 12;
    int k = (s & 7) + ((L >> 4) << 3) + ((s >> 3) << 4);
    Wd_sw[i] = (bf16)Wd[(kk * 32 + k) * 128 + t * 16 + (L & 15)];
  } else if (i < 2 * 32 * 8 * 32 * 16) {   // W_up: 4 ksteps x 64 c-tiles
    int j = i - 32 * 8 * 32 * 16;
    int s = j & 15, L = (j >> 4) & 31, t = (j >> 9) & 63, kk = j >> 15;
    int k = (s & 7) + ((L >> 4) << 3) + ((s >> 3) << 4);
    Wu_sw[j] = (bf16)Wu[(kk * 32 + k) * 1024 + t * 16 + (L & 15)];
  }
}

// ---------------------------------------------------------------------------
// Kernel A: down = gelu(x @ W_down + b_down) -> bf16 row-major; token rows
// also stored d-major (V^T) with packed v8bf stores (batch stride 2248 and
// P=200 are multiples of 8, so each lane's 8 accumulator rows are 8
// consecutive same-region tokens).
// 8 waves/WG, wave: 16 rows x 128 cols, K=1024 in 32 steps.
// W_down fragments double-buffered through LDS; all 8 B-fragments preloaded
// per k-step so the 8 WMMAs issue back-to-back after one dscnt wait.
// ---------------------------------------------------------------------------
__global__ __launch_bounds__(256) void k_down_gelu(
    const float* __restrict__ x, const bf16* __restrict__ Wd_sw,
    const float* __restrict__ b_down,
    bf16* __restrict__ down, bf16* __restrict__ Vt) {
  __shared__ __attribute__((aligned(32))) bf16 ldsW[2][4096]; // 2 x 8KB chunks
  const int tid  = threadIdx.x;
  const int lane = tid & 31, wave = tid >> 5;
  const int h    = lane >> 4;
  const int rowbase = blockIdx.x * 128 + wave * 16;
  const int arow    = rowbase + (lane & 15);

  // Preload chunk 0 into buffer 0.
  *(v16bf*)&ldsW[0][tid * 16] = *(const v16bf*)&Wd_sw[tid * 16];

  v8f acc[8] = {};
  for (int kk = 0; kk < 32; ++kk) {
    __syncthreads();                       // chunk kk visible, buf kk^1 free
    const int cb = kk & 1;
    // Prefetch next chunk into registers (hides under the 8 WMMAs).
    v16bf nxt;
    const bool have = (kk + 1) < 32;
    if (have) nxt = *(const v16bf*)&Wd_sw[(kk + 1) * 4096 + tid * 16];
    // A fragment from x (fp32 -> bf16 on the fly).
    const float* xp = x + (size_t)arow * 1024 + kk * 32 + 8 * h;
    float4 f0 = ((const float4*)xp)[0];
    float4 f1 = ((const float4*)xp)[1];
    float4 f2 = ((const float4*)(xp + 16))[0];
    float4 f3 = ((const float4*)(xp + 16))[1];
    // All 8 B fragments up front -> single dscnt wait before the WMMA burst.
    v16bf bfrag[8];
#pragma unroll
    for (int t = 0; t < 8; ++t)
      bfrag[t] = *(const v16bf*)&ldsW[cb][(t * 32 + lane) * 16];
    v16bf a;
    a[0]=(bf16)f0.x; a[1]=(bf16)f0.y; a[2]=(bf16)f0.z; a[3]=(bf16)f0.w;
    a[4]=(bf16)f1.x; a[5]=(bf16)f1.y; a[6]=(bf16)f1.z; a[7]=(bf16)f1.w;
    a[8]=(bf16)f2.x; a[9]=(bf16)f2.y; a[10]=(bf16)f2.z; a[11]=(bf16)f2.w;
    a[12]=(bf16)f3.x; a[13]=(bf16)f3.y; a[14]=(bf16)f3.z; a[15]=(bf16)f3.w;
#pragma unroll
    for (int t = 0; t < 8; ++t)
      acc[t] = WMMA_BF16(a, bfrag[t], acc[t]);
    if (have) *(v16bf*)&ldsW[cb ^ 1][tid * 16] = nxt;
  }
  // Epilogue: bias + exact gelu; row-major bf16 scatter + packed V^T stores.
  const int flat0 = rowbase + 8 * h;       // 8 consecutive rows per lane-half
  const int bb = flat0 / N_, n0 = flat0 % N_;
  const bool isTok = (n0 >= P_);
  bf16* vb = Vt + (size_t)(bb * 128) * 2048 + (n0 - P_);
#pragma unroll
  for (int t = 0; t < 8; ++t) {
    const int col = t * 16 + (lane & 15);
    const float bias = b_down[col];
    v8bf pack;
#pragma unroll
    for (int r = 0; r < 8; ++r) {
      float v = acc[t][r] + bias;
      v = 0.5f * v * (1.0f + erff(v * 0.70710678118654752f));
      const bf16 bv = (bf16)v;
      down[(size_t)(flat0 + r) * 128 + col] = bv;
      pack[r] = bv;
    }
    if (isTok) *(v8bf*)(vb + (size_t)col * 2048) = pack; // 8 tokens, 16B store
  }
}

// ---------------------------------------------------------------------------
// Kernel B: flash cross-attention. One wave per (batch, 16-row q-tile).
// Computes S^T = K·Q^T and O^T = V^T·P^T so softmax state is per-lane
// (prompt = lane&15) and the exp(S)->P^T fragment needs one shfl per reg.
// Writes prompt_out in place over prompt rows of `down` ("combined").
// ---------------------------------------------------------------------------
__global__ __launch_bounds__(32) void k_attn(bf16* __restrict__ downAll,
                                             const bf16* __restrict__ VtAll) {
  const int lane = threadIdx.x & 31;
  const int h    = lane >> 4;
  const int q0   = blockIdx.x * 16;            // 13 tiles cover P=200 (masked)
  const int b    = blockIdx.y;
  bf16* down = downAll + (size_t)b * N_ * 128;
  const bf16* tokp = down + (size_t)P_ * 128;
  const bf16* Vt   = VtAll + (size_t)b * 128 * 2048;

  // Q^T B-fragments: lane = prompt col, slots = d (8 at 8h, 8 at 16+8h).
  v16bf qf[4];
  {
    const bf16* qrow = down + (size_t)(q0 + (lane & 15)) * 128;
#pragma unroll
    for (int kk = 0; kk < 4; ++kk)
      qf[kk] = cat8(*(const v8bf*)(qrow + kk * 32 + 8 * h),
                    *(const v8bf*)(qrow + kk * 32 + 8 * h + 16));
  }

  float m = -INFINITY, l = 0.f;
  v8f ot[8] = {};
  for (int j0 = 0; j0 < T_; j0 += 32) {
    v8f st0 = {}, st1 = {};
    const bf16* k0row = tokp + (size_t)(j0 + (lane & 15)) * 128 + 8 * h;
    const bf16* k1row = k0row + 16 * 128;
#pragma unroll
    for (int kk = 0; kk < 4; ++kk) {
      v16bf a0 = cat8(*(const v8bf*)(k0row + kk * 32),
                      *(const v8bf*)(k0row + kk * 32 + 16));
      st0 = WMMA_BF16(a0, qf[kk], st0);
      v16bf a1 = cat8(*(const v8bf*)(k1row + kk * 32),
                      *(const v8bf*)(k1row + kk * 32 + 16));
      st1 = WMMA_BF16(a1, qf[kk], st1);
    }
    // online softmax over the 32-token chunk (lane pair L, L^16 = one prompt)
    float e0[8], e1[8], cmax = -INFINITY;
#pragma unroll
    for (int r = 0; r < 8; ++r) {
      e0[r] = st0[r] * SCALE_;
      e1[r] = st1[r] * SCALE_;
      cmax = fmaxf(cmax, fmaxf(e0[r], e1[r]));
    }
    cmax = fmaxf(cmax, __shfl_xor(cmax, 16, 32));
    const float mn = fmaxf(m, cmax);
    const float alpha = __expf(m - mn);
    float s = 0.f;
#pragma unroll
    for (int r = 0; r < 8; ++r) {
      e0[r] = __expf(e0[r] - mn);
      e1[r] = __expf(e1[r] - mn);
      s += e0[r] + e1[r];
    }
    s += __shfl_xor(s, 16, 32);
    l = l * alpha + s;
    m = mn;
#pragma unroll
    for (int t = 0; t < 8; ++t)
#pragma unroll
      for (int r = 0; r < 8; ++r) ot[t][r] *= alpha;
    // Build P^T B-fragment: slots 0..7 = tokens 16h+0..7, 8..15 = 16h+8..15.
    v16bf pb;
#pragma unroll
    for (int r = 0; r < 8; ++r) {
      const float send = h ? e0[r] : e1[r];
      const float recv = __shfl_xor(send, 16, 32);
      pb[r]     = (bf16)(h ? recv  : e0[r]);
      pb[8 + r] = (bf16)(h ? e1[r] : recv);
    }
    // O^T += V^T · P^T   (A-fragments contiguous thanks to d-major Vt)
#pragma unroll
    for (int t = 0; t < 8; ++t) {
      const bf16* vp = Vt + (size_t)(t * 16 + (lane & 15)) * 2048 + j0 + 16 * h;
      v16bf va = cat8(*(const v8bf*)vp, *(const v8bf*)(vp + 8));
      ot[t] = WMMA_BF16(va, pb, ot[t]);
    }
  }
  const float inv = 1.0f / l;
  const int p = lane & 15;
  if (q0 + p < P_) {
    bf16* orow = down + (size_t)(q0 + p) * 128;
#pragma unroll
    for (int t = 0; t < 8; ++t)
#pragma unroll
      for (int r = 0; r < 8; ++r)
        orow[t * 16 + r + 8 * h] = (bf16)(ot[t][r] * inv);
  }
}

// ---------------------------------------------------------------------------
// Kernel C: out = gate * (combined @ W_up + b_up), fp32 output.
// 8 waves/WG; each wave computes TWO 16-row tiles x 128 cols (8 c-tiles),
// sharing every B-fragment across both row tiles in registers -> halves the
// W_up_sw L2 streaming (~0.6 GB total) and B-load instructions per WMMA.
// ---------------------------------------------------------------------------
__global__ __launch_bounds__(256) void k_up(const bf16* __restrict__ comb,
                                            const bf16* __restrict__ Wu_sw,
                                            const float* __restrict__ b_up,
                                            const float* __restrict__ gate,
                                            float* __restrict__ out) {
  const int tid = threadIdx.x, lane = tid & 31, wave = tid >> 5;
  const int h = lane >> 4;
  const int rowbase = blockIdx.x * 32;        // 32 rows per workgroup
  const int cg = wave;                        // 8 col-groups of 8 tiles each
  const bf16* arow0 = comb + (size_t)(rowbase + (lane & 15)) * 128 + 8 * h;
  const bf16* arow1 = arow0 + 16 * 128;
  v16bf af0[4], af1[4];
#pragma unroll
  for (int kk = 0; kk < 4; ++kk) {
    af0[kk] = cat8(*(const v8bf*)(arow0 + kk * 32),
                   *(const v8bf*)(arow0 + kk * 32 + 16));
    af1[kk] = cat8(*(const v8bf*)(arow1 + kk * 32),
                   *(const v8bf*)(arow1 + kk * 32 + 16));
  }
  v8f acc0[8] = {}, acc1[8] = {};
#pragma unroll
  for (int kk = 0; kk < 4; ++kk)
#pragma unroll
    for (int t = 0; t < 8; ++t) {
      v16bf bf = *(const v16bf*)&Wu_sw[(((kk * 64) + cg * 8 + t) * 32 + lane) * 16];
      acc0[t] = WMMA_BF16(af0[kk], bf, acc0[t]);
      acc1[t] = WMMA_BF16(af1[kk], bf, acc1[t]);
    }
  const float g = gate[0];
#pragma unroll
  for (int t = 0; t < 8; ++t) {
    const int col = (cg * 8 + t) * 16 + (lane & 15);
    const float bias = b_up[col];
#pragma unroll
    for (int r = 0; r < 8; ++r) {
      out[(size_t)(rowbase + r + 8 * h) * 1024 + col]      = g * (acc0[t][r] + bias);
      out[(size_t)(rowbase + 16 + r + 8 * h) * 1024 + col] = g * (acc1[t][r] + bias);
    }
  }
}

// ---------------------------------------------------------------------------
extern "C" void kernel_launch(void* const* d_in, const int* in_sizes, int n_in,
                              void* d_out, int out_size, void* d_ws, size_t ws_size,
                              hipStream_t stream) {
  const float* x    = (const float*)d_in[0];
  const float* Wd   = (const float*)d_in[1];
  const float* bd   = (const float*)d_in[2];
  const float* Wu   = (const float*)d_in[3];
  const float* bu   = (const float*)d_in[4];
  const float* gate = (const float*)d_in[5];

  char* ws = (char*)d_ws;
  bf16* Wd_sw = (bf16*)(ws);                                   // 256 KB
  bf16* Wu_sw = (bf16*)(ws + 262144);                          // 256 KB
  bf16* down  = (bf16*)(ws + 524288);                          // B*N*128 bf16
  bf16* Vt    = (bf16*)(ws + 524288 + (size_t)B_ * N_ * 128 * 2); // B*128*2048

  k_swizzle_weights<<<1024, 256, 0, stream>>>(Wd, Wu, Wd_sw, Wu_sw);
  k_down_gelu<<<562, 256, 0, stream>>>(x, Wd_sw, bd, down, Vt); // 562*128 rows
  k_attn<<<dim3(13, 32), 32, 0, stream>>>(down, Vt);            // 13 qtiles x B
  k_up<<<2248, 256, 0, stream>>>(down, Wu_sw, bu, gate, (float*)d_out);
}
